// DPRNNBase_35510789603699
// MI455X (gfx1250) — matrix operand order repaired and checked by reading, not compiled, not run on hardware
//
#include <hip/hip_runtime.h>

// ---------------------------------------------------------------------------
// 2-layer LSTM for MI455X (gfx1250, wave32, WMMA).
//   T=512, B=64, D=H=1024, 4H=4096.
// bf16 WMMA (16x16x32) with fp32 accumulation; fp32 cell state.
// Weights pre-packed into WMMA-fragment layout; B staged via LDS (shared by
// 4 waves per block, double-buffered, async global->LDS when available);
// A fragments register double-buffered.
// ---------------------------------------------------------------------------

typedef __bf16 bf16;
typedef bf16  v16bf __attribute__((ext_vector_type(16)));
typedef float v8f   __attribute__((ext_vector_type(8)));

#define T_STEPS 512
#define BATCH   64
#define DIM     1024
#define HID     1024
#define G4      4096
#define TC      64      // time-chunk for gx precompute
#define KB      32      // K / 32

// --- optional CDNA5 async global->LDS staging (guarded; fallback = VGPR bounce)
#if defined(__has_builtin)
#  if __has_builtin(__builtin_amdgcn_global_load_async_to_lds_b128) && \
      __has_builtin(__builtin_amdgcn_s_wait_asynccnt)
#    define USE_ASYNC_LDS 1
#  endif
#endif
#ifndef USE_ASYNC_LDS
#  define USE_ASYNC_LDS 0
#endif

#if USE_ASYNC_LDS
#  define ASYNC_FENCE() __builtin_amdgcn_s_wait_asynccnt(0)
typedef int v4i __attribute__((vector_size(16)));
typedef __attribute__((address_space(1))) v4i as1_v4i;
typedef __attribute__((address_space(3))) v4i as3_v4i;
#else
#  define ASYNC_FENCE() ((void)0)
#endif

__device__ __forceinline__ float sigf(float x) {
    return 1.0f / (1.0f + __expf(-x));
}

// --- A fragment from row-major [rows, K] bf16 (contiguous k order, as in k_pack_w) ------
// lanes 0-15: K {0..15}; lanes 16-31: {16..31}
__device__ __forceinline__ v16bf load_a_frag(const bf16* __restrict__ base, int lane) {
    const uint32_t* p = (const uint32_t*)base;   // pairs of bf16
    int kh2 = (lane < 16) ? 0 : 8;
    union { v16bf v; uint32_t u[8]; } r;
#pragma unroll
    for (int i = 0; i < 8; ++i) {
        int pairIdx = kh2 + i;
        r.u[i] = p[pairIdx];
    }
    return r.v;
}

// --- B fragment from LDS-staged packed chunk -------------------------------
// chunk layout: [gate][lane][16 bf16]  (lane's 32 bytes are contiguous)
__device__ __forceinline__ v16bf lds_b_frag(const char* smem, int gate, int lane) {
    const uint32_t* p = (const uint32_t*)(smem + (size_t)(gate * 32 + lane) * 32);
    union { v16bf v; uint32_t u[8]; } r;
#pragma unroll
    for (int i = 0; i < 8; ++i) r.u[i] = p[i];
    return r.v;
}

// --- cooperative stage of one packed 4KB weight chunk into LDS -------------
// wpack chunk (nTile, kb): 4 gates x 32 lanes x 32B = 4096 B contiguous.
__device__ __forceinline__ void stage_b(const bf16* __restrict__ wpack, char* dst,
                                        int nTile, int kb, int tid) {
#if USE_ASYNC_LDS
    const char* srcb = (const char*)wpack + (size_t)(nTile * KB + kb) * 4096 + (size_t)tid * 32;
    char* dstb = dst + (size_t)tid * 32;
    __builtin_amdgcn_global_load_async_to_lds_b128(
        (as1_v4i*)srcb, (as3_v4i*)dstb, 0, 0);
    __builtin_amdgcn_global_load_async_to_lds_b128(
        (as1_v4i*)(srcb + 16), (as3_v4i*)(dstb + 16), 0, 0);
#else
    const uint4* src = (const uint4*)(wpack + (size_t)((nTile * KB + kb) * 4) * 32 * 16);
    uint4* d = (uint4*)dst;
    d[tid * 2]     = src[tid * 2];
    d[tid * 2 + 1] = src[tid * 2 + 1];
#endif
}

// --- helper kernels --------------------------------------------------------
__global__ void k_cvt_bf16(const float* __restrict__ src, bf16* __restrict__ dst, int n) {
    int i = blockIdx.x * blockDim.x + threadIdx.x;
    if (i < n) dst[i] = (bf16)src[i];
}

__global__ void k_bias_sum(const float* __restrict__ bi, const float* __restrict__ bh,
                           float* __restrict__ bias, int n) {
    int i = blockIdx.x * blockDim.x + threadIdx.x;
    if (i < n) bias[i] = bi[i] + bh[i];
}

__global__ void k_init_hc(const float* __restrict__ h0, const float* __restrict__ c0,
                          bf16* __restrict__ hbuf0, float* __restrict__ cbuf, int n) {
    int i = blockIdx.x * blockDim.x + threadIdx.x;
    if (i < n) { hbuf0[i] = (bf16)h0[i]; cbuf[i] = c0[i]; }
}

// --- pack fp32 weight [4H, K] into WMMA fragment layout (bf16) -------------
// wp[nTile(64)][kb(32)][gate(4)][lane(32)][e(16)]
//   col = nTile*16 + (lane&15); k = kb*32 + (lane<16 ? 0 : 16) + e
__global__ void k_pack_w(const float* __restrict__ w, bf16* __restrict__ wp) {
    int idx = blockIdx.x * blockDim.x + threadIdx.x;   // < 4H*K = 4M
    if (idx >= G4 * DIM) return;
    int e    = idx & 15;
    int lane = (idx >> 4) & 31;
    int gate = (idx >> 9) & 3;
    int kb   = (idx >> 11) & 31;
    int nT   = idx >> 16;
    int col = nT * 16 + (lane & 15);
    int k   = kb * 32 + ((lane < 16) ? 0 : 16) + e;
    wp[idx] = (bf16)w[(size_t)(gate * HID + col) * DIM + k];
}

// --- input projection: gx = In . Wih^T + bias ------------------------------
// Block: 128 thr (4 waves) = 4 consecutive M-tiles sharing one N-tile.
// Grid: (TC*B/64) mGroups * 64 nTiles = 64*64 = 4096 blocks.
__global__ void k_gx_gemm(const bf16* __restrict__ inbf,   // [T*B, DIM]
                          const bf16* __restrict__ wpack,  // packed [4H, DIM]
                          const float* __restrict__ bias,  // [4H]
                          float* __restrict__ gx,          // [TC*B, 4H]
                          int t0) {
    __shared__ __align__(16) char smem[2][4096];
    int tid    = threadIdx.x;
    int waveId = tid >> 5;
    int lane   = tid & 31;
    int nTile  = blockIdx.x & 63;
    int mTile  = (blockIdx.x >> 6) * 4 + waveId;
    int rowA   = mTile * 16 + (lane & 15);
    int col    = nTile * 16 + (lane & 15);

    const bf16* aRow = inbf + (size_t)(t0 * BATCH + rowA) * DIM;
    v8f acc[4] = {};

    stage_b(wpack, smem[0], nTile, 0, tid);
    v16bf a_cur = load_a_frag(aRow, lane);

    for (int kb = 0; kb < KB; ++kb) {
        ASYNC_FENCE();
        __syncthreads();                       // smem[kb&1] ready
        const char* buf = smem[kb & 1];
        v16bf b0 = lds_b_frag(buf, 0, lane);
        v16bf b1 = lds_b_frag(buf, 1, lane);
        v16bf b2 = lds_b_frag(buf, 2, lane);
        v16bf b3 = lds_b_frag(buf, 3, lane);
        v16bf a_nxt = a_cur;
        if (kb + 1 < KB) {
            stage_b(wpack, smem[(kb + 1) & 1], nTile, kb + 1, tid);
            a_nxt = load_a_frag(aRow + (kb + 1) * 32, lane);
        }
        if (kb + 4 < KB) {
            const char* pf = (const char*)(wpack + (size_t)((nTile * KB + kb + 4) * 4) * 32 * 16);
            __builtin_prefetch(pf + tid * 32, 0, 1);
        }
        acc[0] = __builtin_amdgcn_wmma_f32_16x16x32_bf16(false, a_cur, false, b0, (short)0, acc[0], false, false);
        acc[1] = __builtin_amdgcn_wmma_f32_16x16x32_bf16(false, a_cur, false, b1, (short)0, acc[1], false, false);
        acc[2] = __builtin_amdgcn_wmma_f32_16x16x32_bf16(false, a_cur, false, b2, (short)0, acc[2], false, false);
        acc[3] = __builtin_amdgcn_wmma_f32_16x16x32_bf16(false, a_cur, false, b3, (short)0, acc[3], false, false);
        a_cur = a_nxt;
    }

    int rowBase = mTile * 16 + ((lane < 16) ? 0 : 8);
#pragma unroll
    for (int gate = 0; gate < 4; ++gate) {
        float bv = bias[gate * HID + col];
#pragma unroll
        for (int j = 0; j < 8; ++j) {
            int r = rowBase + j;
            gx[(size_t)r * G4 + gate * HID + col] = acc[gate][j] + bv;
        }
    }
}

// --- fused recurrent step: gates = gx_t + h.Whh^T; LSTM cell update --------
// Block: 128 thr (4 waves) = full batch (4 M-tiles) sharing one N-tile.
// Grid: 64 blocks (one per 16-wide hidden strip).
__global__ void k_lstm_step(const float* __restrict__ gx,    // [TC*B, 4H]
                            const bf16* __restrict__ wpack,  // packed [4H, HID]
                            const bf16* __restrict__ hprev,  // [B, HID]
                            bf16* __restrict__ hnext,        // [B, HID]
                            float* __restrict__ cbuf,        // [B, HID]
                            bf16* __restrict__ hbf_out,      // next-layer input row (L0) or null
                            float* __restrict__ ys_out,      // d_out ys row (L1) or null
                            float* __restrict__ hn_out,      // final-h slot or null
                            float* __restrict__ cn_out,      // final-c slot or null
                            int tLocal) {
    __shared__ __align__(16) char smem[2][4096];
    int tid    = threadIdx.x;
    int waveId = tid >> 5;
    int lane   = tid & 31;
    int nTile  = blockIdx.x;     // 0..63
    int mTile  = waveId;         // 0..3
    int rowA   = mTile * 16 + (lane & 15);
    int col    = nTile * 16 + (lane & 15);

    const bf16* aRow = hprev + (size_t)rowA * HID;
    v8f acc[4] = {};

    stage_b(wpack, smem[0], nTile, 0, tid);
    v16bf a_cur = load_a_frag(aRow, lane);

    for (int kb = 0; kb < KB; ++kb) {
        ASYNC_FENCE();
        __syncthreads();
        const char* buf = smem[kb & 1];
        v16bf b0 = lds_b_frag(buf, 0, lane);
        v16bf b1 = lds_b_frag(buf, 1, lane);
        v16bf b2 = lds_b_frag(buf, 2, lane);
        v16bf b3 = lds_b_frag(buf, 3, lane);
        v16bf a_nxt = a_cur;
        if (kb + 1 < KB) {
            stage_b(wpack, smem[(kb + 1) & 1], nTile, kb + 1, tid);
            a_nxt = load_a_frag(aRow + (kb + 1) * 32, lane);
        }
        acc[0] = __builtin_amdgcn_wmma_f32_16x16x32_bf16(false, a_cur, false, b0, (short)0, acc[0], false, false);
        acc[1] = __builtin_amdgcn_wmma_f32_16x16x32_bf16(false, a_cur, false, b1, (short)0, acc[1], false, false);
        acc[2] = __builtin_amdgcn_wmma_f32_16x16x32_bf16(false, a_cur, false, b2, (short)0, acc[2], false, false);
        acc[3] = __builtin_amdgcn_wmma_f32_16x16x32_bf16(false, a_cur, false, b3, (short)0, acc[3], false, false);
        a_cur = a_nxt;
    }

    int rowBase = mTile * 16 + ((lane < 16) ? 0 : 8);
    const float* gxr = gx + (size_t)tLocal * BATCH * G4;

#pragma unroll
    for (int j = 0; j < 8; ++j) {
        int r = rowBase + j;
        size_t gbase = (size_t)r * G4 + col;
        float iv = sigf (acc[0][j] + gxr[gbase]);
        float fv = sigf (acc[1][j] + gxr[gbase + 1 * HID]);
        float gv = tanhf(acc[2][j] + gxr[gbase + 2 * HID]);
        float ov = sigf (acc[3][j] + gxr[gbase + 3 * HID]);

        size_t hidx = (size_t)r * HID + col;
        float cn = fv * cbuf[hidx] + iv * gv;
        float hv = ov * tanhf(cn);
        cbuf[hidx]  = cn;
        hnext[hidx] = (bf16)hv;
        if (hbf_out) hbf_out[hidx] = (bf16)hv;
        if (ys_out)  ys_out[hidx]  = hv;
        if (hn_out)  { hn_out[hidx] = hv; cn_out[hidx] = cn; }
    }
}

// ---------------------------------------------------------------------------
extern "C" void kernel_launch(void* const* d_in, const int* in_sizes, int n_in,
                              void* d_out, int out_size, void* d_ws, size_t ws_size,
                              hipStream_t stream) {
    (void)in_sizes; (void)n_in; (void)out_size; (void)ws_size;

    const float* x   = (const float*)d_in[0];   // [T,B,D]
    const float* h0  = (const float*)d_in[1];   // [L,B,H]
    const float* c0  = (const float*)d_in[2];   // [L,B,H]
    const float* wih = (const float*)d_in[3];   // [L,4H,D]
    const float* whh = (const float*)d_in[4];   // [L,4H,H]
    const float* bih = (const float*)d_in[5];   // [L,4H]
    const float* bhh = (const float*)d_in[6];   // [L,4H]

    float* out    = (float*)d_out;
    float* ys_out = out;                                        // [T,B,H]
    float* hn_out = out + (size_t)T_STEPS * BATCH * HID;        // [L,B,H]
    float* cn_out = hn_out + (size_t)2 * BATCH * HID;           // [L,B,H]

    char* ws = (char*)d_ws;
    size_t off = 0;
    auto alloc = [&](size_t bytes) -> char* {
        char* p = ws + off;
        off = (off + bytes + 255) & ~(size_t)255;
        return p;
    };
    bf16*  inbf  = (bf16*) alloc((size_t)T_STEPS * BATCH * DIM * 2);  // 64 MB (x, then ys0)
    bf16*  wihp  = (bf16*) alloc((size_t)2 * G4 * DIM * 2);           // 16 MB packed
    bf16*  whhp  = (bf16*) alloc((size_t)2 * G4 * HID * 2);           // 16 MB packed
    float* bias  = (float*)alloc((size_t)2 * G4 * 4);
    float* gx    = (float*)alloc((size_t)TC * BATCH * G4 * 4);        // 64 MB
    bf16*  hbuf  = (bf16*) alloc((size_t)2 * BATCH * HID * 2);        // ping-pong h
    float* cbuf  = (float*)alloc((size_t)BATCH * HID * 4);

    // one-time conversions / weight packing
    int nX = T_STEPS * BATCH * DIM;
    k_cvt_bf16<<<(nX + 255) / 256, 256, 0, stream>>>(x, inbf, nX);
    int nWL = G4 * DIM;                                   // per-layer weight elems
    for (int l = 0; l < 2; ++l) {
        k_pack_w<<<(nWL + 255) / 256, 256, 0, stream>>>(wih + (size_t)l * nWL, wihp + (size_t)l * nWL);
        k_pack_w<<<(nWL + 255) / 256, 256, 0, stream>>>(whh + (size_t)l * nWL, whhp + (size_t)l * nWL);
    }
    k_bias_sum<<<(2 * G4 + 255) / 256, 256, 0, stream>>>(bih, bhh, bias, 2 * G4);

    for (int l = 0; l < 2; ++l) {
        int nHC = BATCH * HID;
        k_init_hc<<<(nHC + 255) / 256, 256, 0, stream>>>(
            h0 + (size_t)l * nHC, c0 + (size_t)l * nHC, hbuf /* slot 0 */, cbuf, nHC);

        for (int chunk = 0; chunk < T_STEPS / TC; ++chunk) {
            int t0 = chunk * TC;
            // input projection for this time-chunk (large parallel WMMA GEMM)
            k_gx_gemm<<<4096, 128, 0, stream>>>(
                inbf, wihp + (size_t)l * nWL, bias + (size_t)l * G4, gx, t0);

            // sequential recurrence
            for (int tl = 0; tl < TC; ++tl) {
                int t = t0 + tl;
                bf16* hprev = hbuf + (size_t)(t & 1) * BATCH * HID;
                bf16* hnext = hbuf + (size_t)((t + 1) & 1) * BATCH * HID;
                bf16*  hbf_o = (l == 0) ? (inbf + (size_t)t * BATCH * HID) : nullptr;
                float* ys_o  = (l == 1) ? (ys_out + (size_t)t * BATCH * HID) : nullptr;
                bool last = (t == T_STEPS - 1);
                k_lstm_step<<<64, 128, 0, stream>>>(
                    gx, whhp + (size_t)l * nWL, hprev, hnext, cbuf,
                    hbf_o, ys_o,
                    last ? (hn_out + (size_t)l * BATCH * HID) : nullptr,
                    last ? (cn_out + (size_t)l * BATCH * HID) : nullptr,
                    tl);
            }
        }
    }
}
